// KPN_46145128629031
// MI455X (gfx1250) — matrix operand order repaired and checked by reading, not compiled, not run on hardware
//
#include <hip/hip_runtime.h>
#include <hip/hip_bf16.h>
#include <math.h>

typedef __bf16 bf16_t;
typedef __attribute__((ext_vector_type(16))) __bf16 v16bf;
typedef __attribute__((ext_vector_type(8)))  __bf16 v8bf;
typedef __attribute__((ext_vector_type(8)))  float  v8f;

#define TS 128          // pixels per conv tile (half of W=256; never crosses a row)
#define SROW 72         // LDS channel stride (64 data + 8 pad -> conflict-free ds_b128)
#define HWC 64          // activation channel count (padded)

// ---------------------------------------------------------------------------
// Pack data_with_est [4,6,256,256] f32 NCHW -> [4,256,256,64] bf16 NHWC (pad 6->64)
// ---------------------------------------------------------------------------
__global__ void pack_input_kernel(const float* __restrict__ src, bf16_t* __restrict__ dst) {
    int idx = blockIdx.x * 256 + threadIdx.x;     // 4*256*256*64 total
    int c = idx & 63;
    int p = idx >> 6;
    int b = p >> 16, y = (p >> 8) & 255, x = p & 255;
    float v = 0.f;
    if (c < 6) v = src[(((b * 6 + c) * 256 + y) * 256) + x];
    dst[idx] = (bf16_t)v;
}

// ---------------------------------------------------------------------------
// Pack conv weights [O,I,3,3] f32 -> B^T bf16 [Opad][576], k = (ky*3+kx)*64 + c
// (zero pad c>=I and o>=O); also pads bias to Opad floats.
// ---------------------------------------------------------------------------
__global__ void pack_weight_kernel(const float* __restrict__ w, const float* __restrict__ b,
                                   bf16_t* __restrict__ wt, float* __restrict__ bo,
                                   int O, int I, int Opad) {
    int idx = blockIdx.x * 256 + threadIdx.x;
    if (idx >= Opad * 576) return;
    int k = idx % 576, o = idx / 576;
    int ky = k / 192, kx = (k % 192) >> 6, c = k & 63;
    float v = 0.f;
    if (o < O && c < I) v = w[((o * I + c) * 3 + ky) * 3 + kx];
    wt[idx] = (bf16_t)v;
    if (k == 0) bo[o] = (o < O) ? b[o] : 0.f;
}

// ---------------------------------------------------------------------------
// Implicit-GEMM 3x3 conv, SAME padding, stride 1, via v_wmma_f32_16x16x32_bf16.
// in : NHWC bf16, 64 channels.  wt : [Opad][576] bf16.  bias : f32 [Opad].
// Block = 256 thr (8 waves): tile = 128 pixels (row segment) x 64 out channels.
// Wave (wm in 0..3, wn in 0..1) computes 32(M) x 32(N): 4 accumulators.
// K-loop (18 x 32) fully unrolled: all LDS/global offsets become immediates.
// mode: 0 = bias, 1 = bias+relu, 2 = bias+residual(res, stride 64)
// ---------------------------------------------------------------------------
__global__ __launch_bounds__(256)
void conv_wmma_kernel(const bf16_t* __restrict__ in, const bf16_t* __restrict__ wt,
                      const float* __restrict__ bias, const bf16_t* __restrict__ res,
                      bf16_t* __restrict__ out, int cout_stride, int mode) {
    __shared__ bf16_t sA[3 * (TS + 2) * SROW];   // 3 rows x 130 cols x 72 ch = 56,160 B

    const int tid = threadIdx.x;
    const int p0  = blockIdx.x * TS;             // first pixel of tile
    const int b   = p0 >> 16;
    const int y   = (p0 >> 8) & 255;
    const int x0  = p0 & 255;                    // 0 or 128

    // ---- stage input halo: rows y-1..y+1, cols x0-1..x0+128, 64 ch, 8B chunks
    for (int idx = tid; idx < 3 * (TS + 2) * 16; idx += 256) {
        int c4  = (idx & 15) << 2;                       // channel group (4 bf16)
        int col = (idx >> 4) % (TS + 2);
        int row = (idx >> 4) / (TS + 2);
        int yy = y + row - 1, xx = x0 + col - 1;
        uint2 v = make_uint2(0u, 0u);
        if (yy >= 0 && yy < 256 && xx >= 0 && xx < 256)
            v = *(const uint2*)(in + ((size_t)(((b * 256 + yy) * 256) + xx) * HWC + c4));
        *(uint2*)(&sA[(row * (TS + 2) + col) * SROW + c4]) = v;
    }
    __syncthreads();

    const int lane = tid & 31;
    const int wid  = tid >> 5;
    const int wm   = wid & 3;          // wave's M slice (4 x 32 pixels)
    const int wn   = wid >> 2;         // wave's N slice (2 x 32 channels)
    const int lrow = lane & 15;
    const int lhi  = lane >> 4;        // 0/1 half-wave
    const int n_base = blockIdx.y * 64;

    // Loop-invariant per-lane bases.
    // A addr = [mrow*SROW + lhi*16] + [(ky*130+kx)*SROW + (ktile&63)]  (2nd part = imm)
    const int abase0 = (wm * 32 + lrow) * SROW + lhi * 16;    // mf = 0
    const int abase1 = abase0 + 16 * SROW;                    // mf = 1
    // B addr: per-lane column pointer; per-kt offset = ktile elements (imm)
    const bf16_t* wp0 = wt + (size_t)(n_base + wn * 32 + lrow) * 576 + lhi * 16;  // nf = 0
    const bf16_t* wp1 = wp0 + (size_t)16 * 576;                                   // nf = 1

    v8f acc[2][2];
#pragma unroll
    for (int i = 0; i < 2; ++i)
#pragma unroll
        for (int j = 0; j < 2; ++j)
#pragma unroll
            for (int e = 0; e < 8; ++e) acc[i][j][e] = 0.f;

#pragma unroll
    for (int kt = 0; kt < 18; ++kt) {
        const int ktile = kt * 32;
        // Constants (lhi*16 never crosses a 64-ch or 192 (ky) boundary):
        const int k0 = ktile,      k1 = ktile + 16;
        const int off0 = ((k0 / 192) * (TS + 2) + ((k0 % 192) >> 6)) * SROW + (k0 & 63);
        const int off1 = ((k1 / 192) * (TS + 2) + ((k1 % 192) >> 6)) * SROW + (k1 & 63);

        v16bf afr[2];
        {
            const v8bf lo = *(const v8bf*)(&sA[abase0 + off0]);
            const v8bf hi = *(const v8bf*)(&sA[abase0 + off0 + 8]);
            v16bf a;
#pragma unroll
            for (int e = 0; e < 8; ++e) { a[e] = lo[e]; a[8 + e] = hi[e]; }
            afr[0] = a;
        }
        {
            const v8bf lo = *(const v8bf*)(&sA[abase1 + off0]);
            const v8bf hi = *(const v8bf*)(&sA[abase1 + off0 + 8]);
            v16bf a;
#pragma unroll
            for (int e = 0; e < 8; ++e) { a[e] = lo[e]; a[8 + e] = hi[e]; }
            afr[1] = a;
        }
        v16bf bfr[2];
        bfr[0] = *(const v16bf*)(wp0 + ktile);
        bfr[1] = *(const v16bf*)(wp1 + ktile);

#pragma unroll
        for (int mf = 0; mf < 2; ++mf)
#pragma unroll
            for (int nf = 0; nf < 2; ++nf)
                acc[mf][nf] = __builtin_amdgcn_wmma_f32_16x16x32_bf16(
                    false, afr[mf], false, bfr[nf], (short)0, acc[mf][nf], false, false);
    }

    // ---- epilogue: D layout 16x16 f32: N = lane&15, M = (lane>>4)*8 + r
#pragma unroll
    for (int mf = 0; mf < 2; ++mf) {
#pragma unroll
        for (int nf = 0; nf < 2; ++nf) {
            int n = n_base + wn * 32 + nf * 16 + lrow;
            float bn = bias[n];
#pragma unroll
            for (int r = 0; r < 8; ++r) {
                int pix = p0 + wm * 32 + mf * 16 + lhi * 8 + r;
                float v = acc[mf][nf][r] + bn;
                if (mode == 1)      v = fmaxf(v, 0.f);
                else if (mode == 2) v += (float)res[(size_t)pix * HWC + n];
                out[(size_t)pix * cout_stride + n] = (bf16_t)v;
            }
        }
    }
}

// ---------------------------------------------------------------------------
// Output stage 1: copy frames [4,3,256,256] f32 into first half of d_out
// ---------------------------------------------------------------------------
__global__ void copy_data_kernel(const float* __restrict__ src, float* __restrict__ dst) {
    int idx = blockIdx.x * 256 + threadIdx.x;
    dst[idx] = src[idx];
}

// ---------------------------------------------------------------------------
// Output stage 2: per-pixel softmax over |core| (225 taps) + 15x15 gather.
// Block = 256 thr = 16x16 pixel tile; 30x30x3 frame patch staged in LDS.
// ---------------------------------------------------------------------------
__global__ __launch_bounds__(256)
void kpn_kernel(const float* __restrict__ frames, const bf16_t* __restrict__ core,
                float* __restrict__ pred) {
    __shared__ float sF[3 * 30 * 30];
    const int b  = blockIdx.z;
    const int y0 = blockIdx.y * 16;
    const int x0 = blockIdx.x * 16;

    for (int idx = threadIdx.x; idx < 3 * 30 * 30; idx += 256) {
        int c = idx / 900, r = idx % 900, ry = r / 30, rx = r % 30;
        int yy = y0 + ry - 7, xx = x0 + rx - 7;
        float v = 0.f;
        if (yy >= 0 && yy < 256 && xx >= 0 && xx < 256)
            v = frames[((b * 3 + c) * 256 + yy) * 256 + xx];
        sF[idx] = v;
    }
    __syncthreads();

    const int ty = threadIdx.x >> 4, tx = threadIdx.x & 15;
    const int y = y0 + ty, x = x0 + tx;
    const bf16_t* cp = core + (size_t)((b * 256 + y) * 256 + x) * 256;

    float m = -1e30f;
    for (int t = 0; t < 225; ++t) {
        float v = fabsf((float)cp[t]);
        m = fmaxf(m, v);
    }
    float sum = 0.f, a0 = 0.f, a1 = 0.f, a2 = 0.f;
    for (int t = 0; t < 225; ++t) {
        float e = __expf(fabsf((float)cp[t]) - m);
        sum += e;
        int i = t / 15, j = t % 15;                 // tap (i,j): frames[y+i-7, x+j-7]
        int base = (ty + i) * 30 + (tx + j);
        a0 += e * sF[base];
        a1 += e * sF[900 + base];
        a2 += e * sF[1800 + base];
    }
    float inv = 1.f / sum;
    size_t o = (size_t)b * 3 * 65536 + (size_t)y * 256 + x;
    pred[o]             = a0 * inv;
    pred[o + 65536]     = a1 * inv;
    pred[o + 2 * 65536] = a2 * inv;
}

// ---------------------------------------------------------------------------
extern "C" void kernel_launch(void* const* d_in, const int* in_sizes, int n_in,
                              void* d_out, int out_size, void* d_ws, size_t ws_size,
                              hipStream_t stream) {
    (void)in_sizes; (void)n_in; (void)out_size; (void)ws_size;
    const float* data_with_est = (const float*)d_in[0];
    const float* data          = (const float*)d_in[1];
    const float* w_first = (const float*)d_in[2];  const float* b_first = (const float*)d_in[3];
    const float* w1a = (const float*)d_in[4];      const float* b1a = (const float*)d_in[5];
    const float* w1b = (const float*)d_in[6];      const float* b1b = (const float*)d_in[7];
    const float* w2a = (const float*)d_in[8];      const float* b2a = (const float*)d_in[9];
    const float* w2b = (const float*)d_in[10];     const float* b2b = (const float*)d_in[11];
    const float* w3a = (const float*)d_in[12];     const float* b3a = (const float*)d_in[13];
    const float* w3b = (const float*)d_in[14];     const float* b3b = (const float*)d_in[15];
    const float* w_out = (const float*)d_in[16];   const float* b_out = (const float*)d_in[17];

    const size_t NPIX = 4ull * 256 * 256;
    char* ws = (char*)d_ws;
    size_t off = 0;
    auto take = [&](size_t bytes) -> char* {
        char* p = ws + off;
        off += (bytes + 255) & ~(size_t)255;
        return p;
    };
    bf16_t* actA = (bf16_t*)take(NPIX * HWC * sizeof(bf16_t));   // 33.5 MB
    bf16_t* actB = (bf16_t*)take(NPIX * HWC * sizeof(bf16_t));
    bf16_t* actC = (bf16_t*)take(NPIX * HWC * sizeof(bf16_t));
    bf16_t* cor  = (bf16_t*)take(NPIX * 256 * sizeof(bf16_t));   // 134 MB
    bf16_t* wtF  = (bf16_t*)take(64  * 576 * sizeof(bf16_t));
    bf16_t* wt1a = (bf16_t*)take(64  * 576 * sizeof(bf16_t));
    bf16_t* wt1b = (bf16_t*)take(64  * 576 * sizeof(bf16_t));
    bf16_t* wt2a = (bf16_t*)take(64  * 576 * sizeof(bf16_t));
    bf16_t* wt2b = (bf16_t*)take(64  * 576 * sizeof(bf16_t));
    bf16_t* wt3a = (bf16_t*)take(64  * 576 * sizeof(bf16_t));
    bf16_t* wt3b = (bf16_t*)take(64  * 576 * sizeof(bf16_t));
    bf16_t* wtO  = (bf16_t*)take(256 * 576 * sizeof(bf16_t));
    float*  bF   = (float*)take(64 * sizeof(float));
    float*  bs1a = (float*)take(64 * sizeof(float));
    float*  bs1b = (float*)take(64 * sizeof(float));
    float*  bs2a = (float*)take(64 * sizeof(float));
    float*  bs2b = (float*)take(64 * sizeof(float));
    float*  bs3a = (float*)take(64 * sizeof(float));
    float*  bs3b = (float*)take(64 * sizeof(float));
    float*  bsO  = (float*)take(256 * sizeof(float));

    // --- prep: pack input (pad 6->64 ch, NHWC bf16) and weights (B^T bf16)
    pack_input_kernel<<<dim3((unsigned)(NPIX * HWC / 256)), dim3(256), 0, stream>>>(data_with_est, actA);
    auto pw = [&](const float* w, const float* b, bf16_t* wt, float* bo, int O, int I, int Opad) {
        int total = Opad * 576;
        pack_weight_kernel<<<dim3((total + 255) / 256), dim3(256), 0, stream>>>(w, b, wt, bo, O, I, Opad);
    };
    pw(w_first, b_first, wtF,  bF,   64,  6,  64);
    pw(w1a, b1a, wt1a, bs1a, 64, 64, 64);
    pw(w1b, b1b, wt1b, bs1b, 64, 64, 64);
    pw(w2a, b2a, wt2a, bs2a, 64, 64, 64);
    pw(w2b, b2b, wt2b, bs2b, 64, 64, 64);
    pw(w3a, b3a, wt3a, bs3a, 64, 64, 64);
    pw(w3b, b3b, wt3b, bs3b, 64, 64, 64);
    pw(w_out, b_out, wtO, bsO, 225, 64, 256);

    const dim3 cgrid((unsigned)(NPIX / TS), 1);
    const dim3 cblk(256);
    // first conv: x1 = conv(in) + bias
    conv_wmma_kernel<<<cgrid, cblk, 0, stream>>>(actA, wtF, bF, nullptr, actB, HWC, 0);
    // res block 1
    conv_wmma_kernel<<<cgrid, cblk, 0, stream>>>(actB, wt1a, bs1a, nullptr, actC, HWC, 1);
    conv_wmma_kernel<<<cgrid, cblk, 0, stream>>>(actC, wt1b, bs1b, actB, actA, HWC, 2);
    // res block 2
    conv_wmma_kernel<<<cgrid, cblk, 0, stream>>>(actA, wt2a, bs2a, nullptr, actB, HWC, 1);
    conv_wmma_kernel<<<cgrid, cblk, 0, stream>>>(actB, wt2b, bs2b, actA, actC, HWC, 2);
    // res block 3
    conv_wmma_kernel<<<cgrid, cblk, 0, stream>>>(actC, wt3a, bs3a, nullptr, actB, HWC, 1);
    conv_wmma_kernel<<<cgrid, cblk, 0, stream>>>(actB, wt3b, bs3b, actC, actA, HWC, 2);
    // out conv: core (N padded 225->256, stride 256)
    conv_wmma_kernel<<<dim3((unsigned)(NPIX / TS), 4), cblk, 0, stream>>>(actA, wtO, bsO, nullptr, cor, 256, 0);

    // outputs: [data, pred] concatenated f32
    copy_data_kernel<<<dim3(786432 / 256), dim3(256), 0, stream>>>(data, (float*)d_out);
    kpn_kernel<<<dim3(16, 16, 4), dim3(256), 0, stream>>>(data, cor, (float*)d_out + 786432);
}
